// CRF_model_50044958933112
// MI455X (gfx1250) — matrix operand order repaired and checked by reading, not compiled, not run on hardware
//
#include <hip/hip_runtime.h>

#define N_RULES 4095
#define N_TAGS  4096
#define T_STEPS 1024
#define NEGV    (-10000.0f)
#define NBLK    128
#define NTHR    256
#define NTILES  (N_TAGS / 16)    // 256 row-tiles of 16 rows
#define PER_THR (N_TAGS / NTHR)  // 16 acc entries per thread

typedef _Float16 v8h  __attribute__((ext_vector_type(8)));
typedef _Float16 v16h __attribute__((ext_vector_type(16)));
typedef float    v8f  __attribute__((ext_vector_type(8)));

// ---------------------------------------------------------------------------
// Device-wide sense-reversing barrier (persistent-kernel grid sync).
// ctrl[0] = arrival counter, ctrl[1] = generation. Reset by crf_init each call.
// ---------------------------------------------------------------------------
__device__ __forceinline__ void grid_sync(unsigned* ctrl, unsigned nblk) {
  __syncthreads();
  if (threadIdx.x == 0) {
    __threadfence();
    unsigned g = __hip_atomic_load(&ctrl[1], __ATOMIC_RELAXED, __HIP_MEMORY_SCOPE_AGENT);
    unsigned a = __hip_atomic_fetch_add(&ctrl[0], 1u, __ATOMIC_ACQ_REL, __HIP_MEMORY_SCOPE_AGENT);
    if (a == nblk - 1u) {
      __hip_atomic_store(&ctrl[0], 0u, __ATOMIC_RELAXED, __HIP_MEMORY_SCOPE_AGENT);
      __hip_atomic_store(&ctrl[1], g + 1u, __ATOMIC_RELEASE, __HIP_MEMORY_SCOPE_AGENT);
    } else {
      while (__hip_atomic_load(&ctrl[1], __ATOMIC_ACQUIRE, __HIP_MEMORY_SCOPE_AGENT) == g)
        __builtin_amdgcn_s_sleep(2);
    }
  }
  __syncthreads();
}

// ---------------------------------------------------------------------------
// M[r][j] = exp(transitions[r][j]) in f16 (row 0 zeroed: excluded trans_tail
// row; its output is overwritten with NEG anyway). 32 MB, lives in L2.
// ---------------------------------------------------------------------------
__global__ void crf_build_M(const float* __restrict__ trans, _Float16* __restrict__ M) {
  const int r = blockIdx.x;
  const float* src = trans + (size_t)r * N_TAGS;
  _Float16* dst = M + (size_t)r * N_TAGS;
  for (int j = threadIdx.x; j < N_TAGS; j += blockDim.x)
    dst[j] = (r == 0) ? (_Float16)0.0f : (_Float16)__expf(src[j]);
}

// acc ping-pong buffers + barrier control words (must be re-inited every call:
// kernel_launch must be deterministic and d_ws is not re-poisoned).
__global__ void crf_init(float* accA, float* accB, unsigned* ctrl) {
  const int i = blockIdx.x * blockDim.x + threadIdx.x;
  if (i < N_TAGS) {
    accA[i] = (i == 0) ? 0.0f : NEGV;
    accB[i] = NEGV;
  }
  if (i == 0) { ctrl[0] = 0u; ctrl[1] = 0u; }
}

// Gold path score: sum_t feats[t, (prev-1) mod 4095] + transitions[nxt, prev].
__global__ void crf_gold(const float* __restrict__ feats, const float* __restrict__ trans,
                         const int* __restrict__ tags, float* __restrict__ gold) {
  __shared__ float red[NTHR];
  float s = 0.0f;
  for (int t = threadIdx.x; t < T_STEPS; t += blockDim.x) {
    const int prev = (t == 0) ? 0 : tags[t - 1];
    const int nxt  = tags[t];
    const int e = ((prev - 1) % N_RULES + N_RULES) % N_RULES;  // Python % semantics
    s += feats[(size_t)t * N_RULES + e] + trans[(size_t)nxt * N_TAGS + prev];
  }
  red[threadIdx.x] = s;
  __syncthreads();
  for (int o = NTHR / 2; o > 0; o >>= 1) {
    if (threadIdx.x < o) red[threadIdx.x] += red[threadIdx.x + o];
    __syncthreads();
  }
  if (threadIdx.x == 0) *gold = red[0];
}

// ---------------------------------------------------------------------------
// Persistent forward-algorithm kernel. Per step:
//   1) grid barrier; 2) every block redundantly computes m = max(acc) and
//   p[j] = exp(acc[j]-m) -> f16 in LDS (16 KB read from L2, cheap);
//   3) WMMA GEMV: block owns two 16-row tiles; 8 waves split K=4096; partial 16x16 f32
//   accumulators combined through LDS in fixed order; 4) lanes 0/16 of wave 0
//   finalize: acc'[r] = feat[t][r-1] + m + log(dot), acc'[0] = NEG.
// ---------------------------------------------------------------------------
__global__ void __launch_bounds__(NTHR) crf_forward(
    const float* __restrict__ feats, const _Float16* __restrict__ M,
    float* __restrict__ accA, float* __restrict__ accB,
    const float* __restrict__ gold, unsigned* ctrl, float* __restrict__ out)
{
  __shared__ _Float16 p_sh[N_TAGS];     // 8 KB: p vector in f16
  __shared__ float    red[NTHR];        // 1 KB: block reductions
  __shared__ float    cpart[8][8][32];  // 8 KB: per-wave partial C tiles

  const int tid  = threadIdx.x;
  const int wave = tid >> 5;
  const int lane = tid & 31;
  const unsigned nblk = gridDim.x;

  for (int t = 0; t < T_STEPS; ++t) {
    grid_sync(ctrl, nblk);
    const float* acc = (t & 1) ? accB : accA;
    float*       nxt = (t & 1) ? accA : accB;

    // ---- m = max(acc); p = exp(acc - m) ----
    float av[PER_THR];
    float lm = -3.4e38f;
    #pragma unroll
    for (int i = 0; i < PER_THR; ++i) {
      av[i] = acc[tid + i * NTHR];
      lm = fmaxf(lm, av[i]);
    }
    red[tid] = lm;
    __syncthreads();
    for (int o = NTHR / 2; o > 0; o >>= 1) {
      if (tid < o) red[tid] = fmaxf(red[tid], red[tid + o]);
      __syncthreads();
    }
    const float m = red[0];
    __syncthreads();
    #pragma unroll
    for (int i = 0; i < PER_THR; ++i)
      p_sh[tid + i * NTHR] = (_Float16)__expf(av[i] - m);
    __syncthreads();

    // ---- GEMV: new[r] = sum_j M[r,j] * p[j] via v_wmma_f32_16x16x32_f16 ----
    for (int tile = blockIdx.x; tile < NTILES; tile += nblk) {
      const int r0  = tile << 4;
      const int row = r0 + (lane & 15);
      const _Float16* Mrow = M + (size_t)row * N_TAGS;
      const int hiHalf = lane >> 4;      // 0: lanes 0-15, 1: lanes 16-31
      const int koffA  = hiHalf << 4;    // A: hi lanes hold K+16..K+31
      const int koffB  = hiHalf << 4;    // B: hi lanes hold K+16..K+31

      v8f c = {0.f, 0.f, 0.f, 0.f, 0.f, 0.f, 0.f, 0.f};
      const int kbase = wave * (N_TAGS / 8);   // each wave owns K-range of 512
      #pragma unroll
      for (int kk = 0; kk < (N_TAGS / 8) / 32; ++kk) {   // 16 WMMAs per wave
        const int k0 = kbase + kk * 32;
        // A tile (16x32 f16), contiguous layout: lane m: halfs K k0+koffA..+15
        v8h alo = *(const v8h*)(Mrow + k0 + koffA);
        v8h ahi = *(const v8h*)(Mrow + k0 + koffA + 8);
        v16h a = __builtin_shufflevector(alo, ahi,
                   0, 1, 2, 3, 4, 5, 6, 7, 8, 9, 10, 11, 12, 13, 14, 15);
        // B tile (32x16 f16) = p broadcast across N: low lanes K k0..k0+15,
        // high lanes K k0+16..k0+31 (contiguous k order, same as A)
        v16h b = *(const v16h*)(p_sh + k0 + koffB);
        c = __builtin_amdgcn_wmma_f32_16x16x32_f16(
                false, a, false, b, (short)0, c, false, false);
      }
      #pragma unroll
      for (int i = 0; i < 8; ++i) cpart[wave][i][lane] = c[i];
      __syncthreads();

      if (wave == 0) {
        float s[8];
        #pragma unroll
        for (int i = 0; i < 8; ++i) {
          float a8 = 0.0f;
          #pragma unroll
          for (int w = 0; w < 8; ++w) a8 += cpart[w][i][lane];  // fixed order
          s[i] = a8;
        }
        // C/D layout: VGPR i @ lane 0 -> row r0+i; @ lane 16 -> row r0+8+i
        if ((lane & 15) == 0) {
          const int rbase = r0 + (hiHalf << 3);
          #pragma unroll
          for (int i = 0; i < 8; ++i) {
            const int r = rbase + i;
            nxt[r] = (r == 0) ? NEGV
                   : (feats[(size_t)t * N_RULES + (r - 1)] + m + __logf(s[i]));
          }
        }
      }
      __syncthreads();
    }
  }

  // ---- finalize: Z = logsumexp(acc_final); out = Z - gold ----
  grid_sync(ctrl, nblk);
  if (blockIdx.x == 0) {
    const float* acc = accA;  // T even -> final state back in accA
    float av[PER_THR];
    float lm = -3.4e38f;
    #pragma unroll
    for (int i = 0; i < PER_THR; ++i) {
      av[i] = acc[tid + i * NTHR];
      lm = fmaxf(lm, av[i]);
    }
    red[tid] = lm;
    __syncthreads();
    for (int o = NTHR / 2; o > 0; o >>= 1) {
      if (tid < o) red[tid] = fmaxf(red[tid], red[tid + o]);
      __syncthreads();
    }
    const float mz = red[0];
    __syncthreads();
    float ls = 0.0f;
    #pragma unroll
    for (int i = 0; i < PER_THR; ++i) ls += __expf(av[i] - mz);
    red[tid] = ls;
    __syncthreads();
    for (int o = NTHR / 2; o > 0; o >>= 1) {
      if (tid < o) red[tid] += red[tid + o];
      __syncthreads();
    }
    if (tid == 0) out[0] = (mz + __logf(red[0])) - *gold;
  }
}

// ---------------------------------------------------------------------------
extern "C" void kernel_launch(void* const* d_in, const int* in_sizes, int n_in,
                              void* d_out, int out_size, void* d_ws, size_t ws_size,
                              hipStream_t stream) {
  (void)in_sizes; (void)n_in; (void)out_size; (void)ws_size;
  const float* feats = (const float*)d_in[0];   // (1024, 1, 4095) f32
  const float* trans = (const float*)d_in[1];   // (4096, 4096) f32
  const int*   tags  = (const int*)d_in[2];     // (1024,) i32
  float* out = (float*)d_out;                   // (1,) f32

  char* ws = (char*)d_ws;
  _Float16* M = (_Float16*)ws;                                   // 32 MB exp-table
  size_t off = (size_t)N_TAGS * N_TAGS * sizeof(_Float16);
  float* accA = (float*)(ws + off); off += N_TAGS * sizeof(float);
  float* accB = (float*)(ws + off); off += N_TAGS * sizeof(float);
  float* gold = (float*)(ws + off); off += 256;                  // padded slot
  unsigned* ctrl = (unsigned*)(ws + off);                        // barrier state

  crf_build_M<<<N_TAGS, NTHR, 0, stream>>>(trans, M);
  crf_init<<<(N_TAGS + NTHR - 1) / NTHR, NTHR, 0, stream>>>(accA, accB, ctrl);
  crf_gold<<<1, NTHR, 0, stream>>>(feats, trans, tags, gold);
  crf_forward<<<NBLK, NTHR, 0, stream>>>(feats, M, accA, accB, gold, ctrl, out);
}
